// DGCN_7559142441323
// MI455X (gfx1250) — matrix operand order repaired and checked by reading, not compiled, not run on hardware
//
#include <hip/hip_runtime.h>
#include <hip/hip_bf16.h>

typedef __bf16 bf16;
typedef __attribute__((ext_vector_type(16))) __bf16 v16bf;
typedef __attribute__((ext_vector_type(8)))  __bf16 v8bf;
typedef __attribute__((ext_vector_type(8)))  float  v8f;

#define N_NODES 2048
#define EMBEDD  16
#define DIMIO   32
#define BATCH   32
#define SEQ     12
#define NCOLS   (BATCH * DIMIO * SEQ)   // 12288
#define KIO     3072                    // (K+1)*32*32

// CDNA5 async global->LDS copy (16B per lane), tracked by ASYNCcnt.
__device__ __forceinline__ void async_copy_b128(unsigned lds_off, const bf16* gptr) {
  asm volatile("global_load_async_to_lds_b128 %0, %1, off"
               :: "v"(lds_off), "v"(gptr) : "memory");
}
__device__ __forceinline__ void wait_asynccnt0() {
  asm volatile("s_wait_asynccnt 0" ::: "memory");
}

__device__ __forceinline__ v16bf ld_frag(const bf16* p0, const bf16* p1) {
  v16bf f;
  *(v8bf*)&f         = *(const v8bf*)p0;
  *(((v8bf*)&f) + 1) = *(const v8bf*)p1;
  return f;
}

// ---------------- tiny MLP stage 1: h[i,n] = tanh(dot(x[0,i,n,:], fc0_w)+b)
__global__ void k_h(const float* __restrict__ x, const float* __restrict__ fc0w,
                    const float* __restrict__ fc0b, float* __restrict__ h) {
  int idx = blockIdx.x * 256 + threadIdx.x;          // 32*2048 total
  const float* xp = x + (size_t)idx * SEQ;           // x[0,i,n,:]
  float s = 0.f;
  #pragma unroll
  for (int t = 0; t < SEQ; ++t) s += xp[t] * fc0w[t];
  h[idx] = tanhf(s + fc0b[0]);
}

// ---------------- stage 2: h2[i,e] = tanh(h[i,:] . fc1_w[e,:] + fc1_b[e])
__global__ void k_h2(const float* __restrict__ h, const float* __restrict__ fc1w,
                     const float* __restrict__ fc1b, float* __restrict__ h2) {
  int i = blockIdx.x >> 4, e = blockIdx.x & 15;
  int t = threadIdx.x;                                // 64 threads
  float s = 0.f;
  for (int nn = t; nn < N_NODES; nn += 64) s += h[i * N_NODES + nn] * fc1w[e * N_NODES + nn];
  __shared__ float red[64];
  red[t] = s; __syncthreads();
  for (int o = 32; o > 0; o >>= 1) { if (t < o) red[t] += red[t + o]; __syncthreads(); }
  if (t == 0) h2[i * EMBEDD + e] = tanhf(red[0] + fc1b[e]);
}

// ---------------- stage 3: m[d,e] = tanh(sum_i h2[i,d]*fc2_w[e,i] + fc2_b[e])
__global__ void k_m(const float* __restrict__ h2, const float* __restrict__ fc2w,
                    const float* __restrict__ fc2b, float* __restrict__ mmat) {
  int t = threadIdx.x;                                // 256 = 16*16
  int d = t >> 4, e = t & 15;
  float s = 0.f;
  #pragma unroll
  for (int i = 0; i < DIMIO; ++i) s += h2[i * EMBEDD + d] * fc2w[e * DIMIO + i];
  mmat[d * EMBEDD + e] = tanhf(s + fc2b[e]);
}

// ---------------- support row: softmax(relu((E1 m) E2^T)) -> bf16
__global__ __launch_bounds__(256) void k_support(const float* __restrict__ E1,
                                                 const float* __restrict__ E2,
                                                 const float* __restrict__ mmat,
                                                 bf16* __restrict__ Sbf) {
  int n = blockIdx.x, t = threadIdx.x;
  __shared__ float row[N_NODES];
  __shared__ float red[256];
  float tm[EMBEDD];
  #pragma unroll
  for (int e = 0; e < EMBEDD; ++e) {
    float a = 0.f;
    #pragma unroll
    for (int d = 0; d < EMBEDD; ++d) a += E1[n * EMBEDD + d] * mmat[d * EMBEDD + e];
    tm[e] = a;
  }
  float mx = -1e30f;
  for (int c = t; c < N_NODES; c += 256) {
    float v = 0.f;
    #pragma unroll
    for (int e = 0; e < EMBEDD; ++e) v += tm[e] * E2[c * EMBEDD + e];
    v = fmaxf(v, 0.f);
    row[c] = v;
    mx = fmaxf(mx, v);
  }
  red[t] = mx; __syncthreads();
  for (int o = 128; o > 0; o >>= 1) { if (t < o) red[t] = fmaxf(red[t], red[t + o]); __syncthreads(); }
  mx = red[0]; __syncthreads();
  float sm = 0.f;
  for (int c = t; c < N_NODES; c += 256) { float e = __expf(row[c] - mx); row[c] = e; sm += e; }
  red[t] = sm; __syncthreads();
  for (int o = 128; o > 0; o >>= 1) { if (t < o) red[t] += red[t + o]; __syncthreads(); }
  float inv = 1.f / red[0];
  for (int c = t; c < N_NODES; c += 256) Sbf[(size_t)n * N_NODES + c] = (bf16)(row[c] * inv);
}

// ---------------- pack x [b,i,m,s] -> Xp[m, (b*32+i)*12+s] bf16 (K x N for GEMM)
__global__ void k_pack(const float* __restrict__ x, bf16* __restrict__ Xp) {
  int bi = blockIdx.x;                                 // b*32+i, 0..1023
  const float* src = x + (size_t)bi * N_NODES * SEQ;
  int c0 = bi * SEQ;
  for (int m = threadIdx.x; m < N_NODES; m += 256) {
    const float* sp = src + (size_t)m * SEQ;
    bf16* dp = Xp + (size_t)m * NCOLS + c0;
    #pragma unroll
    for (int s = 0; s < SEQ; ++s) dp[s] = (bf16)sp[s];
  }
}

// ---------------- per-node generated weights (bf16) and bias (f32)
__global__ void k_weights(const float* __restrict__ E1, const float* __restrict__ E2,
                          const float* __restrict__ WP, const float* __restrict__ WP2,
                          const float* __restrict__ BP, const float* __restrict__ BP2,
                          bf16* __restrict__ Wbf, float* __restrict__ biasv) {
  int n = blockIdx.x, t = threadIdx.x;
  __shared__ float e1[EMBEDD], e2[EMBEDD];
  if (t < EMBEDD) { e1[t] = E1[n * EMBEDD + t]; e2[t] = E2[n * EMBEDD + t]; }
  __syncthreads();
  for (int idx = t; idx < KIO; idx += 256) {
    float a = 0.f;
    #pragma unroll
    for (int d = 0; d < EMBEDD; ++d)
      a += e1[d] * WP[d * KIO + idx] + e2[d] * WP2[d * KIO + idx];
    Wbf[(size_t)n * KIO + idx] = (bf16)a;
  }
  if (t < DIMIO) {
    float a = 0.f;
    #pragma unroll
    for (int d = 0; d < EMBEDD; ++d) a += e1[d] * BP[d * DIMIO + t] + e2[d] * BP2[d * DIMIO + t];
    biasv[n * DIMIO + t] = a;
  }
}

// ---------------- bf16 WMMA GEMM: C[M,N] = A[M,K] * B[K,N], row-major, bf16 out
// 256 threads = 8 waves; block tile 64x128, K-step 32, double-buffered LDS.
// A tile staged with CDNA5 async global->LDS (no transpose needed);
// B tile staged via VGPR + transposed ds_store_b16 scatter (Bt[n][k]).
#define BM 64
#define BN 128
#define BK 32
__global__ __launch_bounds__(256) void k_gemm_bf16(const bf16* __restrict__ A,
                                                   const bf16* __restrict__ B,
                                                   bf16* __restrict__ C,
                                                   int Kd, int Nn) {
  __shared__ alignas(16) bf16 As[2][BM][40];    // [m][k], row stride 80B
  __shared__ alignas(16) bf16 Bt[2][BN][40];    // [n][k] transposed
  int tid = threadIdx.x;
  int lane = tid & 31, w = tid >> 5;
  int h = lane >> 4, ln = lane & 15;
  int m0 = blockIdx.y * BM, n0 = blockIdx.x * BN;
  int mt = w >> 1, ntb = (w & 1) * 4;           // wave: 1 A-frag x 4 B-frags
  v8f acc[4] = {{}, {}, {}, {}};

  int ar = tid >> 2, ac = (tid & 3) * 8;        // A: 64 rows x 4 chunks of 8 halves
  int br = tid >> 3, bc = (tid & 7) * 16;       // B: 32 rows x 8 chunks of 16 halves
  const bf16* agp = A + (size_t)(m0 + ar) * Kd + ac;
  const bf16* bgp = B + (size_t)br * Nn + n0 + bc;
  int nkt = Kd / BK;

  // prologue: stage tile 0 -> buffer 0
  async_copy_b128((unsigned)(uintptr_t)&As[0][ar][ac], agp);
  {
    v8bf b0v = *(const v8bf*)bgp;
    v8bf b1v = *(const v8bf*)(bgp + 8);
    #pragma unroll
    for (int j = 0; j < 8; ++j) { Bt[0][bc + j][br] = b0v[j]; Bt[0][bc + 8 + j][br] = b1v[j]; }
  }
  agp += BK; bgp += (size_t)BK * Nn;
  wait_asynccnt0();
  __syncthreads();

  for (int kt = 0; kt < nkt; ++kt) {
    int cur = kt & 1, nxt = cur ^ 1;
    bool hasnext = (kt + 1 < nkt);
    v8bf b0v, b1v;
    if (hasnext) {                               // issue next tile's traffic first
      b0v = *(const v8bf*)bgp;
      b1v = *(const v8bf*)(bgp + 8);
      async_copy_b128((unsigned)(uintptr_t)&As[nxt][ar][ac], agp);
      agp += BK; bgp += (size_t)BK * Nn;
    }
    // preload ALL fragments for this buffer, then issue the WMMA chain
    const bf16* arow = &As[cur][mt * 16 + ln][0];
    v16bf af = ld_frag(arow + h * 8, arow + 16 + h * 8);
    v16bf bfr[4];
    #pragma unroll
    for (int q = 0; q < 4; ++q) {
      const bf16* brow = &Bt[cur][(ntb + q) * 16 + ln][0];
      bfr[q] = ld_frag(brow + h * 8, brow + 16 + h * 8);
    }
    #pragma unroll
    for (int q = 0; q < 4; ++q)
      acc[q] = __builtin_amdgcn_wmma_f32_16x16x32_bf16(false, af, false, bfr[q],
                                                       (short)0, acc[q], false, false);
    if (hasnext) {
      #pragma unroll
      for (int j = 0; j < 8; ++j) { Bt[nxt][bc + j][br] = b0v[j]; Bt[nxt][bc + 8 + j][br] = b1v[j]; }
      wait_asynccnt0();
    }
    __syncthreads();
  }

  #pragma unroll
  for (int q = 0; q < 4; ++q) {
    #pragma unroll
    for (int r = 0; r < 8; ++r) {
      int m = m0 + mt * 16 + r + 8 * h;
      C[(size_t)m * Nn + n0 + (ntb + q) * 16 + ln] = (bf16)acc[q][r];
    }
  }
}

// ---------------- final conv: per (n, batch-half): [192 x 96] @ [96 x 32] + bias
__global__ __launch_bounds__(256) void k_conv(const float* __restrict__ x,
                                              const bf16* __restrict__ Y1,
                                              const bf16* __restrict__ Y2,
                                              const bf16* __restrict__ Wbf,
                                              const float* __restrict__ biasv,
                                              float* __restrict__ out) {
  __shared__ alignas(16) bf16 As[192][104];   // rows (b_local,s), cols (k,i)
  __shared__ alignas(16) bf16 Bt[32][104];    // [o][ki] transposed W_n
  __shared__ float bs[32];
  int n = blockIdx.x, bh = blockIdx.y;        // bh in {0,1}: batch half
  int tid = threadIdx.x;
  // stage W_n transposed (vectorized reads)
  const bf16* wp = Wbf + (size_t)n * KIO;
  for (int cidx = tid * 8; cidx < KIO; cidx += 2048) {
    v8bf wv = *(const v8bf*)(wp + cidx);
    #pragma unroll
    for (int j = 0; j < 8; ++j) { int idx = cidx + j; Bt[idx & 31][idx >> 5] = wv[j]; }
  }
  if (tid < 32) bs[tid] = biasv[n * 32 + tid];
  // k=0 hop (identity): x itself
  for (int p = tid; p < 512; p += 256) {
    int bl = p >> 5, i = p & 31;
    int b = bh * 16 + bl;
    const float* xp = x + ((size_t)(b * 32 + i) * N_NODES + n) * SEQ;
    #pragma unroll
    for (int s = 0; s < SEQ; ++s) As[bl * SEQ + s][i] = (bf16)xp[s];
  }
  // k=1,2 hops (vectorized reads of the contiguous row n of Y1/Y2)
  const bf16* y1 = Y1 + (size_t)n * NCOLS + bh * 6144;
  const bf16* y2 = Y2 + (size_t)n * NCOLS + bh * 6144;
  for (int cidx = tid * 8; cidx < 6144; cidx += 2048) {
    v8bf v1 = *(const v8bf*)(y1 + cidx);
    v8bf v2 = *(const v8bf*)(y2 + cidx);
    #pragma unroll
    for (int j = 0; j < 8; ++j) {
      int idx = cidx + j;
      int bl = idx / 384, rem = idx % 384;
      int i = rem / SEQ, s = rem % SEQ;
      As[bl * SEQ + s][32 + i] = v1[j];
      As[bl * SEQ + s][64 + i] = v2[j];
    }
  }
  __syncthreads();
  int lane = tid & 31, w = tid >> 5;
  int h = lane >> 4, ln = lane & 15;
  for (int st = w * 3; st < w * 3 + 3; ++st) {   // 24 subtiles over 8 waves
    int mt = st >> 1, nt = st & 1;
    // preload all 3 k-step fragment pairs, then chain the WMMAs
    v16bf af[3], bv[3];
    #pragma unroll
    for (int kk = 0; kk < 3; ++kk) {
      const bf16* arow = &As[mt * 16 + ln][kk * 32];
      const bf16* brow = &Bt[nt * 16 + ln][kk * 32];
      af[kk] = ld_frag(arow + h * 8, arow + 16 + h * 8);
      bv[kk] = ld_frag(brow + h * 8, brow + 16 + h * 8);
    }
    v8f acc = {};
    #pragma unroll
    for (int kk = 0; kk < 3; ++kk)
      acc = __builtin_amdgcn_wmma_f32_16x16x32_bf16(false, af[kk], false, bv[kk],
                                                    (short)0, acc, false, false);
    int o = nt * 16 + ln;
    float bias = bs[o];
    #pragma unroll
    for (int r = 0; r < 8; ++r) {
      int m = mt * 16 + r + 8 * h;              // 0..191
      int bl = m / SEQ, s = m % SEQ;
      int b = bh * 16 + bl;
      out[((size_t)(b * 32 + o) * N_NODES + n) * SEQ + s] = acc[r] + bias;
    }
  }
}

extern "C" void kernel_launch(void* const* d_in, const int* in_sizes, int n_in,
                              void* d_out, int out_size, void* d_ws, size_t ws_size,
                              hipStream_t stream) {
  (void)in_sizes; (void)n_in; (void)out_size; (void)ws_size;
  const float* x    = (const float*)d_in[0];
  const float* E1   = (const float*)d_in[1];
  const float* E2   = (const float*)d_in[2];
  const float* WP   = (const float*)d_in[3];
  const float* WP2  = (const float*)d_in[4];
  const float* BP   = (const float*)d_in[5];
  const float* BP2  = (const float*)d_in[6];
  const float* fc0w = (const float*)d_in[7];
  const float* fc0b = (const float*)d_in[8];
  const float* fc1w = (const float*)d_in[9];
  const float* fc1b = (const float*)d_in[10];
  const float* fc2w = (const float*)d_in[11];
  const float* fc2b = (const float*)d_in[12];
  float* out = (float*)d_out;

  char* ws = (char*)d_ws;
  size_t off = 0;
  auto alloc = [&](size_t bytes) -> void* {
    void* p = ws + off;
    off += (bytes + 255) & ~(size_t)255;
    return p;
  };
  float* h    = (float*)alloc((size_t)32 * 2048 * 4);
  float* h2   = (float*)alloc((size_t)32 * 16 * 4);
  float* mmat = (float*)alloc((size_t)16 * 16 * 4);
  bf16*  Sbf  = (bf16*)alloc((size_t)2048 * 2048 * 2);
  bf16*  Xp   = (bf16*)alloc((size_t)2048 * NCOLS * 2);
  bf16*  Y1   = (bf16*)alloc((size_t)2048 * NCOLS * 2);
  bf16*  Wbf  = (bf16*)alloc((size_t)2048 * KIO * 2);
  float* bv   = (float*)alloc((size_t)2048 * 32 * 4);
  bf16*  Y2   = Xp;   // Xp is dead after GEMM1 -> reuse for Y2

  k_h<<<256, 256, 0, stream>>>(x, fc0w, fc0b, h);
  k_h2<<<512, 64, 0, stream>>>(h, fc1w, fc1b, h2);
  k_m<<<1, 256, 0, stream>>>(h2, fc2w, fc2b, mmat);
  k_pack<<<1024, 256, 0, stream>>>(x, Xp);
  k_weights<<<2048, 256, 0, stream>>>(E1, E2, WP, WP2, BP, BP2, Wbf, bv);
  k_support<<<2048, 256, 0, stream>>>(E1, E2, mmat, Sbf);

  dim3 gg(NCOLS / BN, N_NODES / BM);   // (96, 32)
  k_gemm_bf16<<<gg, 256, 0, stream>>>(Sbf, Xp, Y1, N_NODES, NCOLS);   // Y1 = S @ X
  k_gemm_bf16<<<gg, 256, 0, stream>>>(Sbf, Y1, Y2, N_NODES, NCOLS);   // Y2 = S @ Y1

  k_conv<<<dim3(2048, 2), 256, 0, stream>>>(x, Y1, Y2, Wbf, bv, out);
}
